// StudentModel_34978213658906
// MI455X (gfx1250) — matrix operand order repaired and checked by reading, not compiled, not run on hardware
//
#include <hip/hip_runtime.h>
#include <math.h>

typedef __attribute__((ext_vector_type(16))) _Float16 v16h;
typedef __attribute__((ext_vector_type(8)))  float    v8f;

// ---------------------------------------------------------------------------
// CSR construction (graph shared by all 3 layers; built once per launch)
// ---------------------------------------------------------------------------
__global__ void count_kernel(const int* __restrict__ dst, int* __restrict__ cnt, int E) {
    int e = blockIdx.x * blockDim.x + threadIdx.x;
    if (e < E) atomicAdd(&cnt[dst[e]], 1);
}

__global__ __launch_bounds__(1024) void scan_kernel(const int* __restrict__ cnt,
                                                    int* __restrict__ rowptr,
                                                    int* __restrict__ cursor, int n) {
    __shared__ int tmp[1024];
    __shared__ int carry;
    int tid = threadIdx.x;
    if (tid == 0) carry = 0;
    __syncthreads();
    for (int start = 0; start < n; start += 1024) {
        int i = start + tid;
        int v = (i < n) ? cnt[i] : 0;
        tmp[tid] = v;
        __syncthreads();
        for (int off = 1; off < 1024; off <<= 1) {
            int t = (tid >= off) ? tmp[tid - off] : 0;
            __syncthreads();
            tmp[tid] += t;
            __syncthreads();
        }
        int excl = tmp[tid] - v + carry;     // exclusive prefix + running carry
        if (i < n) { rowptr[i] = excl; cursor[i] = excl; }
        __syncthreads();
        if (tid == 0) carry += tmp[1023];
        __syncthreads();
    }
    if (tid == 0) rowptr[n] = carry;
}

__global__ void fill_kernel(const int* __restrict__ src, const int* __restrict__ dst,
                            int* __restrict__ cursor, int* __restrict__ colidx, int E) {
    int e = blockIdx.x * blockDim.x + threadIdx.x;
    if (e < E) {
        int p = atomicAdd(&cursor[dst[e]], 1);
        colidx[p] = src[e];
    }
}

// ---------------------------------------------------------------------------
// GEMM  Y[N,DOUT] = X[N,128] * W[128,DOUT] + bias   via WMMA f16-split
// (hi*hi + hi*lo + lo*hi in f32 accumulators -> ~f32 accuracy)
// One wave computes a 16x32 tile of Y (two 16x16 WMMA tiles sharing A).
// All offsets 32-bit; DOUT compile-time.
// ---------------------------------------------------------------------------
template <int DOUT>
__global__ __launch_bounds__(32) void gemm128_wmma(
    const float* __restrict__ X, const float* __restrict__ W,
    const float* __restrict__ bias, float* __restrict__ Y,
    int nRows)
{
    const int rowBase = blockIdx.x * 16;
    const int colBase = blockIdx.y * 32;
    const int lane = threadIdx.x;
    const int r16  = lane & 15;
    const int hi   = lane >> 4;            // 0 or 1 (half-wave)

    v8f c0 = {0.f, 0.f, 0.f, 0.f, 0.f, 0.f, 0.f, 0.f};
    v8f c1 = {0.f, 0.f, 0.f, 0.f, 0.f, 0.f, 0.f, 0.f};

    int arow = rowBase + r16;
    if (arow >= nRows) arow = nRows - 1;   // clamp (stores are guarded)
    const float* xrow = X + arow * 128;

#pragma unroll
    for (int k0 = 0; k0 < 128; k0 += 32) {
        // A: 16x32 f16 layout.  lane half 'hi': j=0..7 -> K = 8*hi + j,
        //                                       j=8..15 -> K = 16 + 8*hi + (j-8)
        // -> two runs of 8 consecutive floats: vectorized b128 loads.
        const float4* xa = reinterpret_cast<const float4*>(xrow + k0 + 8 * hi);
        const float4* xb = reinterpret_cast<const float4*>(xrow + k0 + 16 + 8 * hi);
        float4 a0 = xa[0], a1 = xa[1];     // K run 1 (8 floats)
        float4 a2 = xb[0], a3 = xb[1];     // K run 2 (8 floats)
        float afv[16] = {a0.x, a0.y, a0.z, a0.w, a1.x, a1.y, a1.z, a1.w,
                         a2.x, a2.y, a2.z, a2.w, a3.x, a3.y, a3.z, a3.w};
        v16h ah, al;
#pragma unroll
        for (int j = 0; j < 16; ++j) {
            _Float16 h1 = (_Float16)afv[j];
            ah[j] = h1;
            al[j] = (_Float16)(afv[j] - (float)h1);
        }
        // B: 32x16 f16 layout — col = lane&15, K = 8*hi + (j&7) + 16*(j>>3).  Two col tiles.
        v16h bh0, bl0, bh1, bl1;
#pragma unroll
        for (int j = 0; j < 16; ++j) {
            int roww = (k0 + (j & 7) + 8 * hi + 16 * (j >> 3)) * DOUT + colBase + r16;
            float b0 = W[roww];
            float b1 = W[roww + 16];
            _Float16 h0 = (_Float16)b0;
            _Float16 h1 = (_Float16)b1;
            bh0[j] = h0; bl0[j] = (_Float16)(b0 - (float)h0);
            bh1[j] = h1; bl1[j] = (_Float16)(b1 - (float)h1);
        }
        c0 = __builtin_amdgcn_wmma_f32_16x16x32_f16(false, ah, false, bh0, (short)0, c0, false, false);
        c0 = __builtin_amdgcn_wmma_f32_16x16x32_f16(false, ah, false, bl0, (short)0, c0, false, false);
        c0 = __builtin_amdgcn_wmma_f32_16x16x32_f16(false, al, false, bh0, (short)0, c0, false, false);
        c1 = __builtin_amdgcn_wmma_f32_16x16x32_f16(false, ah, false, bh1, (short)0, c1, false, false);
        c1 = __builtin_amdgcn_wmma_f32_16x16x32_f16(false, ah, false, bl1, (short)0, c1, false, false);
        c1 = __builtin_amdgcn_wmma_f32_16x16x32_f16(false, al, false, bh1, (short)0, c1, false, false);
    }

    float b0 = bias[colBase + r16];
    float b1 = bias[colBase + 16 + r16];
    // C/D layout: VGPR r holds M = r + 8*hi, N = lane&15
    if (rowBase + 16 <= nRows) {
#pragma unroll
        for (int r = 0; r < 8; ++r) {
            int row = rowBase + r + 8 * hi;
            Y[row * DOUT + colBase + r16]      = c0[r] + b0;
            Y[row * DOUT + colBase + 16 + r16] = c1[r] + b1;
        }
    } else {
#pragma unroll
        for (int r = 0; r < 8; ++r) {
            int row = rowBase + r + 8 * hi;
            if (row < nRows) {
                Y[row * DOUT + colBase + r16]      = c0[r] + b0;
                Y[row * DOUT + colBase + 16 + r16] = c1[r] + b1;
            }
        }
    }
}

// ---------------------------------------------------------------------------
// Fused edge-attention with online softmax. One wave = one (node, head).
// 4 edges in flight (8 lanes per edge, float4 per lane), no atomics.
// ---------------------------------------------------------------------------
template <int HEADS>
__global__ __launch_bounds__(256) void attn_kernel(
    const float* __restrict__ Q, const float* __restrict__ K,
    const float* __restrict__ V, const float* __restrict__ S,
    const int* __restrict__ rowptr, const int* __restrict__ colidx,
    float* __restrict__ Hout, int N, int applyElu)
{
    constexpr int HD = HEADS * 32;
    int wid  = blockIdx.x * (blockDim.x >> 5) + (threadIdx.x >> 5);
    int lane = threadIdx.x & 31;
    int node = wid / HEADS;                 // HEADS is a power of two -> shift
    int head = wid & (HEADS - 1);
    if (node >= N) return;

    int g = lane >> 3;   // edge group 0..3
    int j = lane & 7;    // float4 index within the 32-channel row

    const int rowOff = node * HD + head * 32;          // 32-bit offsets
    float4 q4 = reinterpret_cast<const float4*>(Q + rowOff)[j];
    const float scl = 0.17677669529663687f;            // 1/sqrt(32)
    q4.x *= scl; q4.y *= scl; q4.z *= scl; q4.w *= scl;

    int base = rowptr[node];
    int deg  = rowptr[node + 1] - base;

    float m = -INFINITY, lsum = 0.f;
    float4 acc = {0.f, 0.f, 0.f, 0.f};

    for (int c0 = 0; c0 < deg; c0 += 4) {
        __builtin_prefetch(colidx + base + c0 + 64, 0, 1);   // gfx1250 global_prefetch
        int e = c0 + g;
        bool act = (e < deg);
        int src = act ? colidx[base + e] : 0;
        int soff = src * HD + head * 32;
        float4 k4 = reinterpret_cast<const float4*>(K + soff)[j];
        float4 v4 = reinterpret_cast<const float4*>(V + soff)[j];
        float d = q4.x * k4.x + q4.y * k4.y + q4.z * k4.z + q4.w * k4.w;
        // reduce within the 8-lane group (butterfly -> broadcast)
        d += __shfl_xor(d, 1, 32);
        d += __shfl_xor(d, 2, 32);
        d += __shfl_xor(d, 4, 32);
        float alpha = act ? d : -INFINITY;
        float mn = fmaxf(m, alpha);
        if (mn != -INFINITY) {               // guard -inf - -inf = NaN (empty tail group)
            float scale = __expf(m - mn);    // m==-inf first time -> 0
            float p     = __expf(alpha - mn);
            lsum  = lsum * scale + p;
            acc.x = acc.x * scale + p * v4.x;
            acc.y = acc.y * scale + p * v4.y;
            acc.z = acc.z * scale + p * v4.z;
            acc.w = acc.w * scale + p * v4.w;
            m = mn;
        }
    }

    if (deg > 0) {
        // merge the 4 edge-groups' partial softmax states across lanes
#pragma unroll
        for (int mask = 8; mask <= 16; mask <<= 1) {
            float mo = __shfl_xor(m, mask, 32);
            float lo = __shfl_xor(lsum, mask, 32);
            float ax = __shfl_xor(acc.x, mask, 32);
            float ay = __shfl_xor(acc.y, mask, 32);
            float az = __shfl_xor(acc.z, mask, 32);
            float aw = __shfl_xor(acc.w, mask, 32);
            float mn = fmaxf(m, mo);
            float s1 = (m  == -INFINITY) ? 0.f : __expf(m  - mn);
            float s2 = (mo == -INFINITY) ? 0.f : __expf(mo - mn);
            lsum  = lsum * s1 + lo * s2;
            acc.x = acc.x * s1 + ax * s2;
            acc.y = acc.y * s1 + ay * s2;
            acc.z = acc.z * s1 + az * s2;
            acc.w = acc.w * s1 + aw * s2;
            m = mn;
        }
    }
    float inv = (deg > 0) ? 1.0f / lsum : 0.f;

    if (g == 0) {                           // lanes 0..7 write the 32 channels
        float4 s4 = reinterpret_cast<const float4*>(S + rowOff)[j];
        float4 r;
        r.x = acc.x * inv + s4.x;
        r.y = acc.y * inv + s4.y;
        r.z = acc.z * inv + s4.z;
        r.w = acc.w * inv + s4.w;
        if (applyElu) {
            r.x = r.x > 0.f ? r.x : expm1f(r.x);
            r.y = r.y > 0.f ? r.y : expm1f(r.y);
            r.z = r.z > 0.f ? r.z : expm1f(r.z);
            r.w = r.w > 0.f ? r.w : expm1f(r.w);
        }
        reinterpret_cast<float4*>(Hout + rowOff)[j] = r;
    }
}

// ---------------------------------------------------------------------------
// Launcher
// ---------------------------------------------------------------------------
extern "C" void kernel_launch(void* const* d_in, const int* in_sizes, int n_in,
                              void* d_out, int out_size, void* d_ws, size_t ws_size,
                              hipStream_t stream) {
    (void)n_in; (void)out_size; (void)ws_size;

    const float* x = (const float*)d_in[0];
    const int N = in_sizes[0] / 128;

    // params leaves, pytree (sorted-key) order per layer: Wk,Wq,Ws,Wv,bk,bq,bs,bv
    auto Wl = [&](int layer, int idx) { return (const float*)d_in[1 + layer * 8 + idx]; };

    const int* ei = (const int*)d_in[25];
    const int  E  = in_sizes[25] / 2;
    const int* srcArr = ei;
    const int* dstArr = ei + E;

    // workspace layout
    size_t NF = (size_t)N * 128;
    char* w = (char*)d_ws;
    float* Qb = (float*)w; w += NF * sizeof(float);
    float* Kb = (float*)w; w += NF * sizeof(float);
    float* Vb = (float*)w; w += NF * sizeof(float);
    float* Sb = (float*)w; w += NF * sizeof(float);
    float* Hb = (float*)w; w += NF * sizeof(float);
    int* rowptr = (int*)w; w += (size_t)(N + 1) * sizeof(int);
    int* cursor = (int*)w; w += (size_t)N * sizeof(int);
    int* cnt    = (int*)w; w += (size_t)N * sizeof(int);
    int* colidx = (int*)w; w += (size_t)E * sizeof(int);

    // ---- build CSR by destination (once; reused by all 3 layers) ----
    hipMemsetAsync(cnt, 0, (size_t)N * sizeof(int), stream);
    count_kernel<<<(E + 255) / 256, 256, 0, stream>>>(dstArr, cnt, E);
    scan_kernel<<<1, 1024, 0, stream>>>(cnt, rowptr, cursor, N);
    fill_kernel<<<(E + 255) / 256, 256, 0, stream>>>(srcArr, dstArr, cursor, colidx, E);

    // order in flattened params: 0=Wk 1=Wq 2=Ws 3=Wv 4=bk 5=bq 6=bs 7=bv
    auto run_layer4 = [&](const float* Hin, int layer, float* Hout) {   // heads=4, elu
        dim3 gg((N + 15) / 16, 4);                                       // DOUT=128, 32-wide col tiles
        gemm128_wmma<128><<<gg, 32, 0, stream>>>(Hin, Wl(layer, 1), Wl(layer, 5), Qb, N);
        gemm128_wmma<128><<<gg, 32, 0, stream>>>(Hin, Wl(layer, 0), Wl(layer, 4), Kb, N);
        gemm128_wmma<128><<<gg, 32, 0, stream>>>(Hin, Wl(layer, 3), Wl(layer, 7), Vb, N);
        gemm128_wmma<128><<<gg, 32, 0, stream>>>(Hin, Wl(layer, 2), Wl(layer, 6), Sb, N);
        int waves = N * 4;
        attn_kernel<4><<<(waves + 7) / 8, 256, 0, stream>>>(Qb, Kb, Vb, Sb, rowptr, colidx,
                                                            Hout, N, 1);
    };

    run_layer4(x,  0, Hb);     // layer 1 (concat, elu)
    run_layer4(Hb, 1, Hb);     // layer 2 (in-place safe: GEMMs consume Hb before attn writes)

    {   // layer 3: heads=1, DOUT=32, mean==identity, no elu, writes d_out
        dim3 gg((N + 15) / 16, 1);
        gemm128_wmma<32><<<gg, 32, 0, stream>>>(Hb, Wl(2, 1), Wl(2, 5), Qb, N);
        gemm128_wmma<32><<<gg, 32, 0, stream>>>(Hb, Wl(2, 0), Wl(2, 4), Kb, N);
        gemm128_wmma<32><<<gg, 32, 0, stream>>>(Hb, Wl(2, 3), Wl(2, 7), Vb, N);
        gemm128_wmma<32><<<gg, 32, 0, stream>>>(Hb, Wl(2, 2), Wl(2, 6), Sb, N);
        int waves = N;
        attn_kernel<1><<<(waves + 7) / 8, 256, 0, stream>>>(Qb, Kb, Vb, Sb, rowptr, colidx,
                                                            (float*)d_out, N, 0);
    }
}
